// ProteinTransformerAF3_47519518163048
// MI455X (gfx1250) — matrix operand order repaired and checked by reading, not compiled, not run on hardware
//
#include <hip/hip_runtime.h>
#include <hip/hip_bf16.h>
#include <cstdint>

// ---------------------------------------------------------------------------
// Types for CDNA5 WMMA (gfx1250, wave32)
// ---------------------------------------------------------------------------
typedef __attribute__((ext_vector_type(16))) __bf16 v16bf;
typedef __attribute__((ext_vector_type(8)))  float  v8f;

union BF16x16 { uint32_t u[8]; v16bf v; };

__device__ __forceinline__ uint16_t f2bf(float f) {
    uint32_t u = __builtin_bit_cast(uint32_t, f);
    uint32_t r = u + 0x7FFFu + ((u >> 16) & 1u);   // round-to-nearest-even
    return (uint16_t)(r >> 16);
}

__device__ __forceinline__ float sigm(float x) {
    return 1.0f / (1.0f + __expf(-x));
}

__device__ __forceinline__ float wsum32(float v) {
    #pragma unroll
    for (int m = 16; m; m >>= 1) v += __shfl_xor(v, m, 32);
    return v;
}

// CDNA5 async global->LDS copy, 16 bytes per lane. GV mode (SADDR = off).
// LDS destination = low 32 bits of the generic pointer (aperture truncation).
__device__ __forceinline__ void async_copy16(uint32_t lds_off, uint64_t gaddr) {
    asm volatile("global_load_async_to_lds_b128 %0, %1, off"
                 :: "v"(lds_off), "v"(gaddr) : "memory");
}
__device__ __forceinline__ void async_wait0() {
    asm volatile("s_wait_asynccnt 0x0" ::: "memory");
}

// ---------------------------------------------------------------------------
// Generic bf16 GEMM: C[MxN] = A[MxK] @ B[KxN] (+bias), A/B row-major bf16,
// C row-major f32.  Block tile: 64 rows x 128 cols, 8 waves; each wave keeps
// 4 accumulators (4 M-subtiles x its 16-col slice) and reuses one B fragment
// across them -> 4 WMMA per wave per K-step of 32.
// A tile staged via GLOBAL_LOAD_ASYNC_TO_LDS_B128; B staged transposed via DS.
// ---------------------------------------------------------------------------
__global__ __launch_bounds__(256) void gemm_bf16_kernel(
    const uint16_t* __restrict__ A, const uint16_t* __restrict__ B,
    float* __restrict__ C, const float* __restrict__ bias,
    int M, int N, int K)
{
    __shared__ uint16_t sA[64 * 32];     // [row][k] row-major, rows 16B-aligned
    __shared__ uint16_t sB[128 * 34];    // transposed [n][k], padded

    const int tid  = threadIdx.x;
    const int wave = tid >> 5;
    const int lane = tid & 31;
    const int l16  = lane & 15;
    const int lh   = lane >> 4;
    const int m0   = blockIdx.y * 64;
    const int n0   = blockIdx.x * 128;

    const v8f z = {0.f, 0.f, 0.f, 0.f, 0.f, 0.f, 0.f, 0.f};
    v8f acc[4] = {z, z, z, z};

    // A-stage assignment: 64 rows x 4 chunks of 16B -> 256 threads
    const int ar = tid >> 2;             // row 0..63
    const int ac = (tid & 3) * 8;        // element col 0,8,16,24
    const uint32_t ldsA = (uint32_t)(uintptr_t)(&sA[ar * 32 + ac]);

    for (int k0 = 0; k0 < K; k0 += 32) {
        // --- stage A tile 64x32 asynchronously (16B per lane) ---
        async_copy16(ldsA, (uint64_t)(uintptr_t)(A + (size_t)(m0 + ar) * K + k0 + ac));

        // --- stage B tile 32x128 transposed: 2048 dwords, 8 per thread ---
        #pragma unroll
        for (int i = 0; i < 8; ++i) {
            int idx = tid * 8 + i;          // 0..2047
            int kr  = idx >> 6;             // k row 0..31
            int jp  = idx & 63;             // col pair 0..63
            uint32_t u = *(const uint32_t*)(B + (size_t)(k0 + kr) * N + n0 + jp * 2);
            sB[(jp * 2 + 0) * 34 + kr] = (uint16_t)(u & 0xffffu);
            sB[(jp * 2 + 1) * 34 + kr] = (uint16_t)(u >> 16);
        }
        async_wait0();
        __syncthreads();

        // --- one B fragment per wave, reused by 4 M-subtiles ---
        BF16x16 bm;
        #pragma unroll
        for (int v = 0; v < 8; ++v) {
            int kb = ((v >> 2) << 4) + (lh << 3) + ((v & 3) << 1);
            bm.u[v] = *(const uint32_t*)(&sB[(wave * 16 + l16) * 34 + kb]);
        }
        #pragma unroll
        for (int s = 0; s < 4; ++s) {
            BF16x16 af;
            #pragma unroll
            for (int v = 0; v < 8; ++v) {
                int ka = ((v >> 2) << 4) + (lh << 3) + ((v & 3) << 1);
                af.u[v] = *(const uint32_t*)(&sA[(s * 16 + l16) * 32 + ka]);
            }
            acc[s] = __builtin_amdgcn_wmma_f32_16x16x32_bf16(
                         false, af.v, false, bm.v, (short)0, acc[s], false, false);
        }
        __syncthreads();
    }

    const int col = n0 + wave * 16 + l16;
    const float bv = bias ? bias[col] : 0.f;
    #pragma unroll
    for (int s = 0; s < 4; ++s) {
        #pragma unroll
        for (int v = 0; v < 8; ++v) {
            int row = m0 + s * 16 + (lh << 3) + v;
            C[(size_t)row * N + col] = acc[s][v] + bv;
        }
    }
}

// ---------------------------------------------------------------------------
// Fused windowed attention: one block per (window I, batch bi).
// 32 queries x 128 keys x 4 heads, DH=32. Scores + pair bias + mask +
// softmax + P@V per head iteration in LDS; WMMA for QK^T and P@V.
// qkv layout per row: [q(128) | k(128) | v(128)] bf16, q/k layer-normed.
// ---------------------------------------------------------------------------
__global__ __launch_bounds__(256) void attn_kernel(
    const uint16_t* __restrict__ qkv, const float* __restrict__ pair,
    const float* __restrict__ pmask, const float* __restrict__ pw,
    uint16_t* __restrict__ obf)
{
    __shared__ uint16_t sQ[32 * 32];     // [q][d] (16B-aligned rows for async)
    __shared__ uint16_t sK[128 * 34];    // [key][d] padded
    __shared__ uint16_t sVt[32 * 130];   // [d][key]
    __shared__ uint16_t sP[32 * 130];    // probs bf16 [q][key]
    __shared__ float    sS[32 * 129];    // scores f32 [q][key]

    const int tid   = threadIdx.x;
    const int wave  = tid >> 5;
    const int lane  = tid & 31;
    const int l16   = lane & 15;
    const int lh    = lane >> 4;
    const int I     = blockIdx.x;        // window index 0..255
    const int bi    = blockIdx.y;        // batch
    const int rbase = bi * 8192 + I * 32;
    const float scale = 0.17677669529663687f;   // 1/sqrt(32)

    // prefetch the block's pair_rep slab (dominant HBM stream)
    {
        const float* pp = pair + (size_t)((bi * 256 + I) * 32) * 128 * 16 + (size_t)tid * 256;
        __builtin_prefetch(pp, 0, 1);
    }

    for (int h = 0; h < 4; ++h) {
        // --- load Q (32x32 bf16) via async copy: 32 rows x 4 chunks of 16B ---
        if (tid < 128) {
            int r  = tid >> 2;
            int ch = (tid & 3) * 8;
            uint32_t lq = (uint32_t)(uintptr_t)(&sQ[r * 32 + ch]);
            async_copy16(lq, (uint64_t)(uintptr_t)(qkv + (size_t)(rbase + r) * 384 + h * 32 + ch));
        }
        // --- load K (key-major) and V (transposed) with window padding ---
        #pragma unroll
        for (int i = 0; i < 8; ++i) {
            int idx = tid * 8 + i;          // 0..2047
            int j = idx >> 4, dp = idx & 15;
            int p = I * 32 + j - 48;        // pad = (128-32)/2
            bool ok = (p >= 0) && (p < 8192);
            size_t ro = (size_t)(bi * 8192 + (ok ? p : 0)) * 384;
            uint32_t ku = ok ? *(const uint32_t*)(qkv + ro + 128 + h * 32 + dp * 2) : 0u;
            uint32_t vu = ok ? *(const uint32_t*)(qkv + ro + 256 + h * 32 + dp * 2) : 0u;
            *(uint32_t*)(&sK[j * 34 + dp * 2]) = ku;
            sVt[(dp * 2 + 0) * 130 + j] = (uint16_t)(vu & 0xffffu);
            sVt[(dp * 2 + 1) * 130 + j] = (uint16_t)(vu >> 16);
        }
        async_wait0();
        __syncthreads();

        // --- scores S = Q K^T * scale : 16 tiles of 16x16, 2 per wave ---
        #pragma unroll
        for (int t2 = 0; t2 < 2; ++t2) {
            int tt = wave + t2 * 8;
            int mt = tt >> 3, nt = tt & 7;
            BF16x16 af, bm;
            #pragma unroll
            for (int v = 0; v < 8; ++v) {
                int ka = ((v >> 2) << 4) + (lh << 3) + ((v & 3) << 1);
                af.u[v] = *(const uint32_t*)(&sQ[(mt * 16 + l16) * 32 + ka]);
                int kb = ((v >> 2) << 4) + (lh << 3) + ((v & 3) << 1);
                bm.u[v] = *(const uint32_t*)(&sK[(nt * 16 + l16) * 34 + kb]);
            }
            v8f acc = {0.f, 0.f, 0.f, 0.f, 0.f, 0.f, 0.f, 0.f};
            acc = __builtin_amdgcn_wmma_f32_16x16x32_bf16(
                      false, af.v, false, bm.v, (short)0, acc, false, false);
            #pragma unroll
            for (int v = 0; v < 8; ++v) {
                int q = mt * 16 + (lh << 3) + v;
                int j = nt * 16 + l16;
                sS[q * 129 + j] = acc[v] * scale;
            }
        }
        __syncthreads();

        // --- add pair bias (single pass over pair_rep) + mask ---
        for (int i = 0; i < 16; ++i) {
            int e = tid * 16 + i;           // 0..4095
            int q = e >> 7, j = e & 127;
            size_t po = (size_t)((bi * 256 + I) * 32 + q) * 128 + j;
            const float* pv = pair + po * 16;
            float bs = 0.f;
            #pragma unroll
            for (int pp = 0; pp < 16; ++pp) bs += pv[pp] * pw[pp * 4 + h];
            float s = sS[q * 129 + j] + bs;
            sS[q * 129 + j] = (pmask[po] > 0.f) ? s : -1e9f;
        }
        __syncthreads();

        // --- softmax per query row (8 lanes per row, wave32 shfl) ---
        {
            int q = tid >> 3, sub = tid & 7;
            float mx = -3.4e38f;
            #pragma unroll
            for (int i = 0; i < 16; ++i) mx = fmaxf(mx, sS[q * 129 + sub * 16 + i]);
            #pragma unroll
            for (int m = 4; m; m >>= 1) mx = fmaxf(mx, __shfl_xor(mx, m, 8));
            float ev[16];
            float sum = 0.f;
            #pragma unroll
            for (int i = 0; i < 16; ++i) {
                ev[i] = __expf(sS[q * 129 + sub * 16 + i] - mx);
                sum += ev[i];
            }
            #pragma unroll
            for (int m = 4; m; m >>= 1) sum += __shfl_xor(sum, m, 8);
            float inv = 1.f / sum;
            #pragma unroll
            for (int i = 0; i < 16; ++i)
                sP[q * 130 + sub * 16 + i] = f2bf(ev[i] * inv);
        }
        __syncthreads();

        // --- O = P @ V : 4 tiles of 16x16 (M=32 x N=32), waves 0..3 ---
        if (wave < 4) {
            int mt = wave >> 1, nt = wave & 1;
            v8f acc = {0.f, 0.f, 0.f, 0.f, 0.f, 0.f, 0.f, 0.f};
            #pragma unroll
            for (int kc = 0; kc < 4; ++kc) {
                BF16x16 af, bm;
                #pragma unroll
                for (int v = 0; v < 8; ++v) {
                    int ka = kc * 32 + ((v >> 2) << 4) + (lh << 3) + ((v & 3) << 1);
                    af.u[v] = *(const uint32_t*)(&sP[(mt * 16 + l16) * 130 + ka]);
                    int kb = kc * 32 + ((v >> 2) << 4) + (lh << 3) + ((v & 3) << 1);
                    bm.u[v] = *(const uint32_t*)(&sVt[(nt * 16 + l16) * 130 + kb]);
                }
                acc = __builtin_amdgcn_wmma_f32_16x16x32_bf16(
                          false, af.v, false, bm.v, (short)0, acc, false, false);
            }
            #pragma unroll
            for (int v = 0; v < 8; ++v) {
                int q = mt * 16 + (lh << 3) + v;
                int d = nt * 16 + l16;
                obf[(size_t)(rbase + q) * 128 + h * 32 + d] = f2bf(acc[v]);
            }
        }
        __syncthreads();
    }
}

// ---------------------------------------------------------------------------
// LayerNorm prep (wave per row): xres = x*m, xn = LN(xres), lncond = LN(cond),
// cn1 = bf16(lncond*cw + cb), condbf = bf16(cond)
// ---------------------------------------------------------------------------
__global__ __launch_bounds__(256) void prep_ln1_kernel(
    const float* __restrict__ x, const float* __restrict__ cond,
    const float* __restrict__ mask,
    const float* __restrict__ cw, const float* __restrict__ cb,
    float* __restrict__ xres, float* __restrict__ xn,
    float* __restrict__ lncond, uint16_t* __restrict__ cnbf,
    uint16_t* __restrict__ condbf)
{
    int row  = blockIdx.x * 8 + (threadIdx.x >> 5);
    int lane = threadIdx.x & 31;
    float m = mask[row];
    float xv[4], cv[4];
    #pragma unroll
    for (int i = 0; i < 4; ++i) {
        int c = i * 32 + lane;
        xv[i] = x[(size_t)row * 128 + c] * m;
        cv[i] = cond[(size_t)row * 128 + c];
    }
    float s = 0.f, s2 = 0.f, t = 0.f, t2 = 0.f;
    #pragma unroll
    for (int i = 0; i < 4; ++i) { s += xv[i]; s2 += xv[i] * xv[i]; t += cv[i]; t2 += cv[i] * cv[i]; }
    s = wsum32(s); s2 = wsum32(s2); t = wsum32(t); t2 = wsum32(t2);
    float mx = s * (1.f / 128.f),  vx = s2 * (1.f / 128.f) - mx * mx;
    float mc = t * (1.f / 128.f),  vc = t2 * (1.f / 128.f) - mc * mc;
    float rx = rsqrtf(vx + 1e-5f), rc = rsqrtf(vc + 1e-5f);
    #pragma unroll
    for (int i = 0; i < 4; ++i) {
        int c = i * 32 + lane;
        size_t idx = (size_t)row * 128 + c;
        xres[idx] = xv[i];
        xn[idx]   = (xv[i] - mx) * rx;
        float lc  = (cv[i] - mc) * rc;
        lncond[idx] = lc;
        cnbf[idx]   = f2bf(lc * cw[c] + cb[c]);
        condbf[idx] = f2bf(cv[i]);
    }
}

// adaLN epilogue: out_bf16 = (xn * sigmoid(g + gb) + bwo) * mask
__global__ __launch_bounds__(256) void adaln_ep_kernel(
    const float* __restrict__ gcat, const float* __restrict__ xn,
    const float* __restrict__ gb, const float* __restrict__ mask,
    uint16_t* __restrict__ out)
{
    size_t idx = (size_t)blockIdx.x * 256 + threadIdx.x;
    int row = (int)(idx >> 7), c = (int)(idx & 127);
    float g  = gcat[(size_t)row * 256 + c] + gb[c];
    float bw = gcat[(size_t)row * 256 + 128 + c];
    out[idx] = f2bf((xn[idx] * sigm(g) + bw) * mask[row]);
}

// qkv post: bias already added in GEMM; LN q,k over DH=32; convert to bf16
__global__ __launch_bounds__(256) void qkv_post_kernel(
    const float* __restrict__ qf, uint16_t* __restrict__ qn)
{
    int gid = blockIdx.x * 256 + threadIdx.x;   // 0..M*4-1
    int r = gid >> 2, h = gid & 3;
    size_t base = (size_t)r * 384 + h * 32;
    #pragma unroll
    for (int sct = 0; sct < 2; ++sct) {          // q then k: layer-norm
        size_t o = base + sct * 128;
        float vv[32], s = 0.f, s2 = 0.f;
        #pragma unroll
        for (int d = 0; d < 32; ++d) { float v = qf[o + d]; vv[d] = v; s += v; s2 += v * v; }
        float mean = s * (1.f / 32.f);
        float var  = s2 * (1.f / 32.f) - mean * mean;
        float rstd = rsqrtf(var + 1e-5f);
        #pragma unroll
        for (int d = 0; d < 32; ++d) qn[o + d] = f2bf((vv[d] - mean) * rstd);
    }
    {
        size_t o = base + 256;                   // v: straight convert
        #pragma unroll
        for (int d = 0; d < 32; ++d) qn[o + d] = f2bf(qf[o + d]);
    }
}

// residual 1 + LN for second adaLN (wave per row)
__global__ __launch_bounds__(256) void resid1_kernel(
    const float* __restrict__ oproj, const float* __restrict__ g1,
    const float* __restrict__ s1b, const float* __restrict__ xres,
    const float* __restrict__ mask, const float* __restrict__ lncond,
    const float* __restrict__ a2cw, const float* __restrict__ a2cb,
    float* __restrict__ x2, float* __restrict__ xn2, uint16_t* __restrict__ cn2)
{
    int row  = blockIdx.x * 8 + (threadIdx.x >> 5);
    int lane = threadIdx.x & 31;
    float m = mask[row];
    float v[4];
    #pragma unroll
    for (int i = 0; i < 4; ++i) {
        int c = i * 32 + lane;
        size_t idx = (size_t)row * 128 + c;
        float a = oproj[idx];                       // includes bo
        float g = sigm(g1[idx] + s1b[c]);
        float xv = (a * g * m + xres[idx]) * m;
        v[i] = xv;
        x2[idx] = xv;
        cn2[idx] = f2bf(lncond[idx] * a2cw[c] + a2cb[c]);
    }
    float s = 0.f, s2 = 0.f;
    #pragma unroll
    for (int i = 0; i < 4; ++i) { s += v[i]; s2 += v[i] * v[i]; }
    s = wsum32(s); s2 = wsum32(s2);
    float mean = s * (1.f / 128.f);
    float var  = s2 * (1.f / 128.f) - mean * mean;
    float rstd = rsqrtf(var + 1e-5f);
    #pragma unroll
    for (int i = 0; i < 4; ++i) {
        int c = i * 32 + lane;
        xn2[(size_t)row * 128 + c] = (v[i] - mean) * rstd;
    }
}

// transition gate: hb = bf16(silu(h1) * h2), h12 row = [h1(512) | h2(512)]
__global__ __launch_bounds__(256) void trans_mul_kernel(
    const float* __restrict__ h12, uint16_t* __restrict__ hb)
{
    size_t idx = (size_t)blockIdx.x * 256 + threadIdx.x;   // M*512
    int row = (int)(idx >> 9), j = (int)(idx & 511);
    float h1 = h12[(size_t)row * 1024 + j];
    float h2 = h12[(size_t)row * 1024 + 512 + j];
    hb[idx] = f2bf(h1 * sigm(h1) * h2);
}

// final: out = (t*m*sigmoid(g2+s2b)*m + x2)*m
__global__ __launch_bounds__(256) void final_kernel(
    const float* __restrict__ t, const float* __restrict__ g2,
    const float* __restrict__ s2b, const float* __restrict__ x2,
    const float* __restrict__ mask, float* __restrict__ out)
{
    size_t idx = (size_t)blockIdx.x * 256 + threadIdx.x;
    int row = (int)(idx >> 7), c = (int)(idx & 127);
    float m = mask[row];
    float tv = t[idx] * m;
    float g = sigm(g2[idx] + s2b[c]);
    out[idx] = (tv * g * m + x2[idx]) * m;
}

// pack/concat f32 weights -> bf16: dst row k = [s0 row | s1 row | s2 row]
__global__ __launch_bounds__(256) void pack_kernel(
    uint16_t* __restrict__ dst, const float* __restrict__ s0,
    const float* __restrict__ s1, const float* __restrict__ s2,
    int K, int n0, int n1, int n2)
{
    int nt = n0 + n1 + n2;
    int idx = blockIdx.x * 256 + threadIdx.x;
    if (idx >= K * nt) return;
    int k = idx / nt, c = idx % nt;
    float v;
    if (c < n0)            v = s0[(size_t)k * n0 + c];
    else if (c < n0 + n1)  v = s1[(size_t)k * n1 + (c - n0)];
    else                   v = s2[(size_t)k * n2 + (c - n0 - n1)];
    dst[idx] = f2bf(v);
}

// concat f32 bias vectors
__global__ __launch_bounds__(256) void packf_kernel(
    float* __restrict__ dst, const float* __restrict__ s0,
    const float* __restrict__ s1, const float* __restrict__ s2,
    int n0, int n1, int n2)
{
    int idx = blockIdx.x * 256 + threadIdx.x;
    if (idx >= n0 + n1 + n2) return;
    if (idx < n0)            dst[idx] = s0[idx];
    else if (idx < n0 + n1)  dst[idx] = s1[idx - n0];
    else                     dst[idx] = s2[idx - n0 - n1];
}

// ---------------------------------------------------------------------------
extern "C" void kernel_launch(void* const* d_in, const int* in_sizes, int n_in,
                              void* d_out, int out_size, void* d_ws, size_t ws_size,
                              hipStream_t stream)
{
    (void)n_in; (void)out_size; (void)ws_size;
    const float* x      = (const float*)d_in[0];
    const float* pair   = (const float*)d_in[1];
    const float* cond   = (const float*)d_in[2];
    const float* mask   = (const float*)d_in[3];
    const float* pmask  = (const float*)d_in[4];
    const float* a1_cw  = (const float*)d_in[5];
    const float* a1_cb  = (const float*)d_in[6];
    const float* a1_gw  = (const float*)d_in[7];
    const float* a1_gb  = (const float*)d_in[8];
    const float* a1_bw  = (const float*)d_in[9];
    const float* wq     = (const float*)d_in[10];
    const float* bq     = (const float*)d_in[11];
    const float* wk     = (const float*)d_in[12];
    const float* bk     = (const float*)d_in[13];
    const float* wv     = (const float*)d_in[14];
    const float* bv     = (const float*)d_in[15];
    const float* pw     = (const float*)d_in[16];
    const float* wo     = (const float*)d_in[17];
    const float* bo     = (const float*)d_in[18];
    const float* s1w    = (const float*)d_in[19];
    const float* s1b    = (const float*)d_in[20];
    const float* a2_cw  = (const float*)d_in[21];
    const float* a2_cb  = (const float*)d_in[22];
    const float* a2_gw  = (const float*)d_in[23];
    const float* a2_gb  = (const float*)d_in[24];
    const float* a2_bw  = (const float*)d_in[25];
    const float* tw1    = (const float*)d_in[26];
    const float* tw2    = (const float*)d_in[27];
    const float* two_w  = (const float*)d_in[28];
    const float* s2w    = (const float*)d_in[29];
    const float* s2b    = (const float*)d_in[30];

    const int M  = in_sizes[0] / 128;   // 16384 rows
    const int NB = 256;                 // windows per batch (8192/32)
    const int BC = M / 8192;            // batch count

    // ----- workspace arena -----
    size_t off = 0;
    auto alloc = [&](size_t bytes) -> void* {
        void* p = (char*)d_ws + off;
        off = (off + bytes + 255) & ~(size_t)255;
        return p;
    };
    uint16_t* Wcat1 = (uint16_t*)alloc(128 * 256 * 2);
    uint16_t* Wqkv  = (uint16_t*)alloc(128 * 384 * 2);
    uint16_t* WOb   = (uint16_t*)alloc(128 * 128 * 2);
    uint16_t* S1Wb  = (uint16_t*)alloc(128 * 128 * 2);
    uint16_t* Wcat2 = (uint16_t*)alloc(128 * 256 * 2);
    uint16_t* WT12  = (uint16_t*)alloc(128 * 1024 * 2);
    uint16_t* TWOb  = (uint16_t*)alloc(512 * 128 * 2);
    uint16_t* S2Wb  = (uint16_t*)alloc(128 * 128 * 2);
    float*    BQKV  = (float*)alloc(384 * 4);

    float*    xres   = (float*)alloc((size_t)M * 128 * 4);
    float*    xn     = (float*)alloc((size_t)M * 128 * 4);   // reused as xn2
    float*    lnc    = (float*)alloc((size_t)M * 128 * 4);
    float*    x2     = (float*)alloc((size_t)M * 128 * 4);
    uint16_t* cnbf   = (uint16_t*)alloc((size_t)M * 128 * 2); // cn1 then cn2
    uint16_t* condbf = (uint16_t*)alloc((size_t)M * 128 * 2);
    uint16_t* xabf   = (uint16_t*)alloc((size_t)M * 128 * 2); // xa then xt
    uint16_t* obf    = (uint16_t*)alloc((size_t)M * 128 * 2);
    uint16_t* qkvn   = (uint16_t*)alloc((size_t)M * 384 * 2);
    uint16_t* hb     = (uint16_t*)alloc((size_t)M * 512 * 2);
    float*    BIG    = (float*)alloc((size_t)M * 1024 * 4);  // staged f32 scratch

    auto gemm = [&](const uint16_t* A, const uint16_t* B, float* C,
                    const float* bias, int N, int K) {
        dim3 g(N / 128, M / 64);
        gemm_bf16_kernel<<<g, dim3(256), 0, stream>>>(A, B, C, bias, M, N, K);
    };
    auto pack = [&](uint16_t* dst, const float* s0, const float* s1,
                    const float* s2, int K, int n0, int n1, int n2) {
        int tot = K * (n0 + n1 + n2);
        pack_kernel<<<dim3((tot + 255) / 256), dim3(256), 0, stream>>>(
            dst, s0, s1, s2, K, n0, n1, n2);
    };

    // ----- weight packing (f32 -> bf16, concatenated) -----
    pack(Wcat1, a1_gw, a1_bw, nullptr, 128, 128, 128, 0);
    pack(Wqkv,  wq,    wk,    wv,      128, 128, 128, 128);
    pack(WOb,   wo,    nullptr, nullptr, 128, 128, 0, 0);
    pack(S1Wb,  s1w,   nullptr, nullptr, 128, 128, 0, 0);
    pack(Wcat2, a2_gw, a2_bw, nullptr, 128, 128, 128, 0);
    pack(WT12,  tw1,   tw2,   nullptr, 128, 512, 512, 0);
    pack(TWOb,  two_w, nullptr, nullptr, 512, 128, 0, 0);
    pack(S2Wb,  s2w,   nullptr, nullptr, 128, 128, 0, 0);
    packf_kernel<<<dim3(2), dim3(256), 0, stream>>>(BQKV, bq, bk, bv, 128, 128, 128);

    // ----- phase 1: adaLN #1 -----
    prep_ln1_kernel<<<dim3(M / 8), dim3(256), 0, stream>>>(
        x, cond, mask, a1_cw, a1_cb, xres, xn, lnc, cnbf, condbf);
    gemm(cnbf, Wcat1, BIG, nullptr, 256, 128);               // [g | bw] = cn @ [gw|bw]
    adaln_ep_kernel<<<dim3(M * 128 / 256), dim3(256), 0, stream>>>(
        BIG, xn, a1_gb, mask, xabf);

    // ----- phase 2: QKV projection + per-head LN -----
    gemm(xabf, Wqkv, BIG, BQKV, 384, 128);                   // qkv f32 (+bias)
    qkv_post_kernel<<<dim3(M * 4 / 256), dim3(256), 0, stream>>>(BIG, qkvn);

    // ----- phase 3: windowed attention (pair bias read once) -----
    attn_kernel<<<dim3(NB, BC), dim3(256), 0, stream>>>(qkvn, pair, pmask, pw, obf);

    // ----- phase 4: output projection + gate + residual + LN #2 -----
    gemm(obf, WOb, BIG, bo, 128, 128);                       // oproj (+bo)
    gemm(condbf, S1Wb, BIG + (size_t)M * 128, nullptr, 128, 128);  // g1
    resid1_kernel<<<dim3(M / 8), dim3(256), 0, stream>>>(
        BIG, BIG + (size_t)M * 128, s1b, xres, mask, lnc, a2_cw, a2_cb,
        x2, xn, cnbf);                                       // xn reused as xn2

    // ----- phase 5: adaLN #2 + transition -----
    gemm(cnbf, Wcat2, BIG, nullptr, 256, 128);
    adaln_ep_kernel<<<dim3(M * 128 / 256), dim3(256), 0, stream>>>(
        BIG, xn, a2_gb, mask, xabf);                         // xabf reused as xt
    gemm(xabf, WT12, BIG, nullptr, 1024, 128);               // [h1 | h2]
    trans_mul_kernel<<<dim3(M * 512 / 256), dim3(256), 0, stream>>>(BIG, hb);
    gemm(hb, TWOb, BIG, nullptr, 128, 512);                  // t = hb @ t_wo
    gemm(condbf, S2Wb, BIG + (size_t)M * 128, nullptr, 128, 128);  // g2

    // ----- phase 6: final gate + residual -> d_out -----
    final_kernel<<<dim3(M * 128 / 256), dim3(256), 0, stream>>>(
        BIG, BIG + (size_t)M * 128, s2b, x2, mask, (float*)d_out);
}
